// GAT_20529943675471
// MI455X (gfx1250) — matrix operand order repaired and checked by reading, not compiled, not run on hardware
//
#include <hip/hip_runtime.h>
#include <hip/hip_bf16.h>

// ---------------------------------------------------------------------------
// GAT x2 for MI455X (gfx1250, wave32).
//   Dense transforms -> v_wmma_f32_16x16x32_f16, 5 M-tiles per wave, f16
//   operands pre-staged (B pre-transposed) so fragments load as b128.
//   Edge phase -> ordered-uint atomicMax + atomicAdd segment softmax with
//   deferred normalization (out = S/denom + bias at finalize).
// ---------------------------------------------------------------------------

typedef __attribute__((ext_vector_type(16))) _Float16 v16h;
typedef __attribute__((ext_vector_type(8)))  _Float16 v8h;
typedef __attribute__((ext_vector_type(4)))  _Float16 v4h;
typedef __attribute__((ext_vector_type(8)))  float    v8f;

#define LEAKY(v) ((v) >= 0.f ? (v) : 0.2f * (v))

__device__ __forceinline__ unsigned f2ord(float f) {
  unsigned u = __float_as_uint(f);
  return (u & 0x80000000u) ? ~u : (u | 0x80000000u);
}
__device__ __forceinline__ float ord2f(unsigned s) {
  unsigned u = (s & 0x80000000u) ? (s & 0x7fffffffu) : ~s;
  return __uint_as_float(u);
}

// ---------------- operand staging ----------------
__global__ __launch_bounds__(256)
void cvt_f32_to_f16(const float4* __restrict__ src, _Float16* __restrict__ dst,
                    int n4) {
  const int t = blockIdx.x * blockDim.x + threadIdx.x;
  if (t >= n4) return;
  const float4 v = src[t];
  v4h o = {(_Float16)v.x, (_Float16)v.y, (_Float16)v.z, (_Float16)v.w};
  *(v4h*)(dst + 4 * (size_t)t) = o;
}

// W[K,Nout] f32 -> Wt[Nout,K] f16 (transposed so B fragments load contiguous)
__global__ __launch_bounds__(256)
void cvt_w_transpose(const float* __restrict__ W, _Float16* __restrict__ Wt,
                     int K, int Nout) {
  const int t = blockIdx.x * blockDim.x + threadIdx.x;
  if (t >= K * Nout) return;
  const int k = t / Nout, n = t % Nout;
  Wt[(size_t)n * K + k] = (_Float16)W[t];
}

// ---------------- WMMA GEMM: H[M,Nout] = Xh[M,K] @ Wt[Nout,K]^T ------------
// grid.x = M/(16*MT); blockDim.x = (Nout/16)*32; wave w -> column tile w.
// A 16x32 f16: lane l row=l&15; halves 0..7 -> K=k0+(l>>4)*8+j, 8..15 -> +16.
// B 32x16 f16: lane l col=l&15; halves 0..7 -> K=k0+(l>>4)*8+j, 8..15 -> +16.
// D 16x16 f32: lane l col=l&15; VGPR j     -> row (l>>4)*8+j.
template <int MT>
__global__ __launch_bounds__(256)
void gat_wmma_gemm_f16(const _Float16* __restrict__ Xh,
                       const _Float16* __restrict__ Wt,
                       float* __restrict__ H, int K, int Nout) {
  const int lane = threadIdx.x & 31;
  const int wave = threadIdx.x >> 5;
  const int hi   = lane >> 4;
  const int col  = wave * 16 + (lane & 15);
  const int r0   = blockIdx.x * MT * 16 + (lane & 15);

  v8f acc[MT] = {};
  const _Float16* __restrict__ wrow = Wt + (size_t)col * K;

  for (int k0 = 0; k0 < K; k0 += 32) {
    const int kb = k0 + hi * 8;
    const v8h bl = *(const v8h*)(wrow + kb);
    const v8h bh = *(const v8h*)(wrow + kb + 16);
    const v16h b = __builtin_shufflevector(bl, bh, 0, 1, 2, 3, 4, 5, 6, 7, 8,
                                           9, 10, 11, 12, 13, 14, 15);
    const int ka = k0 + hi * 8;
#pragma unroll
    for (int mt = 0; mt < MT; ++mt) {
      const _Float16* __restrict__ xrow = Xh + (size_t)(r0 + mt * 16) * K;
      const v8h al = *(const v8h*)(xrow + ka);
      const v8h ah = *(const v8h*)(xrow + ka + 16);
      const v16h a = __builtin_shufflevector(al, ah, 0, 1, 2, 3, 4, 5, 6, 7, 8,
                                             9, 10, 11, 12, 13, 14, 15);
      acc[mt] = __builtin_amdgcn_wmma_f32_16x16x32_f16(
          false, a, false, b, (short)0, acc[mt], false, false);
    }
  }

#pragma unroll
  for (int mt = 0; mt < MT; ++mt) {
    const int mbase = blockIdx.x * MT * 16 + mt * 16 + hi * 8;
#pragma unroll
    for (int j = 0; j < 8; ++j)
      H[(size_t)(mbase + j) * Nout + col] = acc[mt][j];
  }
}

// -------- per-(node,head) attention halves: As = <h, a_src>, Ad = <h, a_dst>
__global__ __launch_bounds__(256)
void gat_attn_coeff(int Nn, int heads, const float* __restrict__ H,
                    const float* __restrict__ asrc, const float* __restrict__ adst,
                    float* __restrict__ As, float* __restrict__ Ad) {
  const int t = blockIdx.x * blockDim.x + threadIdx.x;
  if (t >= Nn * heads) return;
  const int n = t / heads, h = t % heads;
  const float* hp = H + (long long)n * heads * 32 + h * 32;
  const float* ap = asrc + h * 32;
  const float* bp = adst + h * 32;
  float s = 0.f, d = 0.f;
#pragma unroll 8
  for (int c = 0; c < 32; ++c) {
    const float v = hp[c];
    s = fmaf(v, ap[c], s);
    d = fmaf(v, bp[c], d);
  }
  As[t] = s;
  Ad[t] = d;
}

// -------- segment max init with self-loop edge value (ordered-uint encoding)
__global__ __launch_bounds__(256)
void gat_init_max(int Nn, int heads, const float* __restrict__ As,
                  const float* __restrict__ Ad, unsigned* __restrict__ Menc) {
  const int t = blockIdx.x * blockDim.x + threadIdx.x;
  if (t >= Nn * heads) return;
  float v = As[t] + Ad[t];
  Menc[t] = f2ord(LEAKY(v));
}

// -------- per-edge segment max
__global__ __launch_bounds__(256)
void gat_edge_max(const long long* __restrict__ ei, int E, int heads,
                  const float* __restrict__ As, const float* __restrict__ Ad,
                  unsigned* __restrict__ Menc) {
  const int t = blockIdx.x * blockDim.x + threadIdx.x;
  if (t >= E * heads) return;
  const int e = t / heads, h = t % heads;
  const long long s = ei[e], d = ei[E + e];
  float v = As[s * heads + h] + Ad[d * heads + h];
  v = LEAKY(v);
  atomicMax(&Menc[d * heads + h], f2ord(v));
}

// -------- init denom + unnormalized sum S with the self-loop contribution
__global__ __launch_bounds__(256)
void gat_init_den_agg(int Nn, int heads, const float* __restrict__ H,
                      const float* __restrict__ As, const float* __restrict__ Ad,
                      const unsigned* __restrict__ Menc,
                      float* __restrict__ den, float* __restrict__ S) {
  const int CH = heads * 32;
  const int t = blockIdx.x * blockDim.x + threadIdx.x;
  if (t >= Nn * CH) return;
  const int n = t / CH, ch = t % CH, h = ch >> 5;
  float v = As[n * heads + h] + Ad[n * heads + h];
  v = LEAKY(v);
  const float w = __expf(v - ord2f(Menc[n * heads + h]));
  S[t] = w * H[t];
  if ((ch & 31) == 0) den[n * heads + h] = w;
}

// -------- per-edge softmax denominator
__global__ __launch_bounds__(256)
void gat_edge_den(const long long* __restrict__ ei, int E, int heads,
                  const float* __restrict__ As, const float* __restrict__ Ad,
                  const unsigned* __restrict__ Menc, float* __restrict__ den) {
  const int t = blockIdx.x * blockDim.x + threadIdx.x;
  if (t >= E * heads) return;
  const int e = t / heads, h = t % heads;
  const long long s = ei[e], d = ei[E + e];
  float v = As[s * heads + h] + Ad[d * heads + h];
  v = LEAKY(v);
  atomicAdd(&den[d * heads + h], __expf(v - ord2f(Menc[d * heads + h])));
}

// -------- per-edge weighted scatter-sum: one wave per edge, CH/32 ch per lane
__global__ __launch_bounds__(256)
void gat_edge_agg(const long long* __restrict__ ei, int E, int heads,
                  const float* __restrict__ H, const float* __restrict__ As,
                  const float* __restrict__ Ad, const unsigned* __restrict__ Menc,
                  float* __restrict__ S) {
  const int wv   = (blockIdx.x * blockDim.x + threadIdx.x) >> 5;
  const int lane = threadIdx.x & 31;
  if (wv >= E) return;
  const long long s = ei[wv], d = ei[E + wv];
  const int CH  = heads * 32;
  const int cpl = CH >> 5;        // channels per lane (4 for heads=4, 1 for 1)
  const int c0  = lane * cpl;
  const int h   = c0 >> 5;        // all cpl channels share one head
  float v = As[s * heads + h] + Ad[d * heads + h];
  v = LEAKY(v);
  const float ex = __expf(v - ord2f(Menc[d * heads + h]));
  const float* __restrict__ hs = H + (long long)s * CH + c0;
  float* __restrict__ op       = S + (long long)d * CH + c0;
  for (int j = 0; j < cpl; ++j) atomicAdd(&op[j], ex * hs[j]);
}

// -------- layer-1 finalize: relu(S/den + bias) written as f16 GEMM operand
__global__ __launch_bounds__(256)
void gat_finalize_relu_f16(int Nn, int heads, const float* __restrict__ den,
                           const float* __restrict__ bias,
                           const float* __restrict__ S,
                           _Float16* __restrict__ O) {
  const int CH = heads * 32;
  const int t = blockIdx.x * blockDim.x + threadIdx.x;
  if (t >= Nn * CH) return;
  const int n = t / CH, ch = t % CH, h = ch >> 5;
  const float v = S[t] / den[n * heads + h] + bias[ch];
  O[t] = (_Float16)fmaxf(v, 0.f);
}

// -------- layer-2 finalize: out = S/den + bias (f32, in place)
__global__ __launch_bounds__(256)
void gat_finalize(int Nn, int heads, const float* __restrict__ den,
                  const float* __restrict__ bias, float* __restrict__ S) {
  const int CH = heads * 32;
  const int t = blockIdx.x * blockDim.x + threadIdx.x;
  if (t >= Nn * CH) return;
  const int n = t / CH, ch = t % CH, h = ch >> 5;
  S[t] = S[t] / den[n * heads + h] + bias[ch];
}

static inline int cdiv(int a, int b) { return (a + b - 1) / b; }

extern "C" void kernel_launch(void* const* d_in, const int* in_sizes, int n_in,
                              void* d_out, int out_size, void* d_ws, size_t ws_size,
                              hipStream_t stream) {
  const float*     x   = (const float*)d_in[0];
  const long long* ei  = (const long long*)d_in[1];   // [2,E] int64: row0=src, row1=dst
  const float*     W1  = (const float*)d_in[2];
  const float*     as1 = (const float*)d_in[3];
  const float*     ad1 = (const float*)d_in[4];
  const float*     b1  = (const float*)d_in[5];
  const float*     W2  = (const float*)d_in[6];
  const float*     as2 = (const float*)d_in[7];
  const float*     ad2 = (const float*)d_in[8];
  const float*     b2  = (const float*)d_in[9];
  float*           out = (float*)d_out;

  const int Nn = in_sizes[0] / 128;   // 50000 (= 625 * 5 * 16)
  const int E  = in_sizes[1] / 2;     // 1.6M

  // ---- workspace carve-up (~62 MB + 40 KB of f16 weights) ----
  float*    h1  = (float*)d_ws;                         // [N,128] layer-1 transform
  float*    As1 = h1 + (size_t)Nn * 128;                // [N,4]
  float*    Ad1 = As1 + (size_t)Nn * 4;                 // [N,4]
  unsigned* M1  = (unsigned*)(Ad1 + (size_t)Nn * 4);    // [N,4] ordered-uint max
  float*    D1  = (float*)(M1 + (size_t)Nn * 4);        // [N,4] denom
  float*    S1  = D1 + (size_t)Nn * 4;                  // [N,128] agg buffer
  float*    hB  = S1 + (size_t)Nn * 128;                // [N,32] layer-2 transform
  float*    As2 = hB + (size_t)Nn * 32;                 // [N]
  float*    Ad2 = As2 + Nn;                             // [N]
  unsigned* M2  = (unsigned*)(Ad2 + Nn);                // [N]
  float*    D2  = (float*)(M2 + Nn);                    // [N]
  _Float16* W1t = (_Float16*)(D2 + Nn);                 // [128,128] f16 (transposed)
  _Float16* W2t = W1t + 128 * 128;                      // [32,128]  f16 (transposed)
  // Aliases (lifetimes do not overlap):
  _Float16* Xh  = (_Float16*)S1;  // f16 copy of x; dead before S1's f32 use
  _Float16* h2h = (_Float16*)h1;  // relu'd layer-1 output; h1 dead by then

  const int B = 256;
  const int g5 = Nn / (16 * 5);   // 625 blocks of 5 row-tiles

  // ---- stage f16 operands ----
  cvt_f32_to_f16<<<cdiv(Nn * 32, B), B, 0, stream>>>((const float4*)x, Xh, Nn * 32);
  cvt_w_transpose<<<cdiv(128 * 128, B), B, 0, stream>>>(W1, W1t, 128, 128);
  cvt_w_transpose<<<cdiv(128 * 32, B), B, 0, stream>>>(W2, W2t, 128, 32);

  // ================= Layer 1 (heads=4, out=32) =================
  gat_wmma_gemm_f16<5><<<g5, 8 * 32, 0, stream>>>(Xh, W1t, h1, 128, 128);
  gat_attn_coeff<<<cdiv(Nn * 4, B), B, 0, stream>>>(Nn, 4, h1, as1, ad1, As1, Ad1);
  gat_init_max<<<cdiv(Nn * 4, B), B, 0, stream>>>(Nn, 4, As1, Ad1, M1);
  gat_edge_max<<<cdiv(E * 4, B), B, 0, stream>>>(ei, E, 4, As1, Ad1, M1);
  gat_init_den_agg<<<cdiv(Nn * 128, B), B, 0, stream>>>(Nn, 4, h1, As1, Ad1, M1, D1, S1);
  gat_edge_den<<<cdiv(E * 4, B), B, 0, stream>>>(ei, E, 4, As1, Ad1, M1, D1);
  gat_edge_agg<<<cdiv(E * 32, B), B, 0, stream>>>(ei, E, 4, h1, As1, Ad1, M1, S1);
  gat_finalize_relu_f16<<<cdiv(Nn * 128, B), B, 0, stream>>>(Nn, 4, D1, b1, S1, h2h);

  // ================= Layer 2 (heads=1, out=32) =================
  gat_wmma_gemm_f16<5><<<g5, 2 * 32, 0, stream>>>(h2h, W2t, hB, 128, 32);
  gat_attn_coeff<<<cdiv(Nn, B), B, 0, stream>>>(Nn, 1, hB, as2, ad2, As2, Ad2);
  gat_init_max<<<cdiv(Nn, B), B, 0, stream>>>(Nn, 1, As2, Ad2, M2);
  gat_edge_max<<<cdiv(E, B), B, 0, stream>>>(ei, E, 1, As2, Ad2, M2);
  gat_init_den_agg<<<cdiv(Nn * 32, B), B, 0, stream>>>(Nn, 1, hB, As2, Ad2, M2, D2, out);
  gat_edge_den<<<cdiv(E, B), B, 0, stream>>>(ei, E, 1, As2, Ad2, M2, D2);
  gat_edge_agg<<<cdiv(E * 32, B), B, 0, stream>>>(ei, E, 1, hB, As2, Ad2, M2, out);
  gat_finalize<<<cdiv(Nn * 32, B), B, 0, stream>>>(Nn, 1, D2, b2, out);
}
